// DABlock_46411416600646
// MI455X (gfx1250) — matrix operand order repaired and checked by reading, not compiled, not run on hardware
//
#include <hip/hip_runtime.h>
#include <hip/hip_bf16.h>

// ---------------- problem constants ----------------
#define BB     8
#define CC     256
#define HH     64
#define WW     64
#define HWSZ   4096          // HH*WW
#define HEADS  4
#define DHEAD  64
#define GROUPS 4
#define GD     64            // CC/GROUPS
#define DSTR   4
#define KOFF   6
#define HP     16
#define WP     16
#define PP     256           // HP*WP
#define BG     32            // BB*GROUPS
#define NELEM  (BB*CC*HWSZ)  // 8388608

typedef __bf16 bf16x16 __attribute__((ext_vector_type(16)));
typedef float  v8f     __attribute__((ext_vector_type(8)));

// ---------------- WMMA helpers ----------------
__device__ __forceinline__ v8f wmma_bf16(bf16x16 a, bf16x16 b, v8f c) {
    return __builtin_amdgcn_wmma_f32_16x16x32_bf16(
        false, a, false, b, (short)0, c, false, false);
}

// A-matrix 16x32 bf16 fragment: lane m = L%16, khalf = L/16.
// elements e<8: K = khalf*8 + e ; e>=8: K = 16 + khalf*8 + (e-8)
__device__ __forceinline__ bf16x16 load_a_frag(const __bf16* rowK0, int khalf) {
    union { bf16x16 v; uint4 u[2]; } f;
    f.u[0] = *(const uint4*)(rowK0 + khalf * 8);
    f.u[1] = *(const uint4*)(rowK0 + 16 + khalf * 8);
    return f.v;
}

// B-matrix 32x16 bf16 fragment: lane n = L%16, khalf = L/16.
// elements e<8: K = khalf*8 + e ; e>=8: K = 16 + khalf*8 + (e-8)
__device__ __forceinline__ bf16x16 load_b_frag(const __bf16* rowK0, int khalf) {
    union { bf16x16 v; uint4 u[2]; } f;
    f.u[0] = *(const uint4*)(rowK0 + khalf * 8);
    f.u[1] = *(const uint4*)(rowK0 + 16 + khalf * 8);
    return f.v;
}

// ---------------- CDNA5 async global<->LDS (ASYNCcnt) ----------------
// per-lane 16B copy: LDS[lds_off] = MEM[gaddr]   (GLOBAL_LOAD_ASYNC_TO_LDS_B128)
__device__ __forceinline__ void async_ld_b128(unsigned lds_off, const void* g) {
    asm volatile("global_load_async_to_lds_b128 %0, %1, off"
                 :: "v"(lds_off), "v"((unsigned long long)(uintptr_t)g)
                 : "memory");
}
// per-lane 16B copy: MEM[gaddr] = LDS[lds_off]   (GLOBAL_STORE_ASYNC_FROM_LDS_B128)
__device__ __forceinline__ void async_st_b128(void* g, unsigned lds_off) {
    asm volatile("global_store_async_from_lds_b128 %0, %1, off"
                 :: "v"((unsigned long long)(uintptr_t)g), "v"(lds_off)
                 : "memory");
}
__device__ __forceinline__ void wait_async0() {
    asm volatile("s_wait_asynccnt 0" ::: "memory");
}
__device__ __forceinline__ unsigned lds_off32(const void* p) {
    return (unsigned)(uintptr_t)p;  // LDS flat aperture: addr[31:0] = LDS byte offset
}

// ---------------- utility kernels ----------------
__global__ void cvt_f32_bf16_kernel(const float* __restrict__ in,
                                    __bf16* __restrict__ out, int n) {
    int i = blockIdx.x * blockDim.x + threadIdx.x;
    int stride = gridDim.x * blockDim.x;
    for (; i < n; i += stride) out[i] = (__bf16)in[i];
}

__global__ void bn_prep_kernel(const float* g1, const float* b1, const float* m1, const float* v1,
                               const float* g2, const float* b2, const float* m2, const float* v2,
                               float* s1, float* sh1, float* s2, float* sh2) {
    int i = blockIdx.x * blockDim.x + threadIdx.x;
    if (i < 2 * CC) { float sc = g1[i] * rsqrtf(v1[i] + 1e-3f); s1[i] = sc; sh1[i] = b1[i] - m1[i] * sc; }
    if (i < CC)     { float sc = g2[i] * rsqrtf(v2[i] + 1e-3f); s2[i] = sc; sh2[i] = b2[i] - m2[i] * sc; }
}

// ---------------- generic batched bf16 GEMM with epilogues ----------------
// out[z][m][n] = epilogue( sum_k W[m][k] * X[z][k][n] )
// MODE 0: out16 = acc
// MODE 1: v = acc + shift[m] + resid;          out32 = v, out16 = v   (Wo + bo + x)
// MODE 2: v = silu(acc*scale[m] + shift[m]);   out16 = v              (f1 BN+SiLU)
// MODE 3: v = acc*scale[m] + shift[m] + resid; out32 = v              (f2 BN + residual)
// W tile staged by async global->LDS copies; X tile (needs transpose) is
// software-pipelined through VGPRs. Both tiles double-buffered.
template <int MODE>
__global__ __launch_bounds__(256) void gemm_bf16_kernel(
    const __bf16* __restrict__ Wt, const __bf16* __restrict__ X,
    float* __restrict__ out32, __bf16* __restrict__ out16,
    const float* __restrict__ scale, const float* __restrict__ shift,
    const float* __restrict__ resid,
    int Mtot, int Ntot, int KDIM) {
    __shared__ __align__(16) __bf16 Wl[2][128 * 40];  // [m][k], row stride 40 (pad)
    __shared__ __align__(16) __bf16 Xl[2][128 * 40];  // [n][k], row stride 40 (pad)

    const int tid  = threadIdx.x;
    const int lane = tid & 31;
    const int wid  = tid >> 5;
    const int wm   = wid & 1;        // wave row block (64 rows)
    const int wn   = wid >> 1;       // wave col block (32 cols)
    const int m0   = blockIdx.y * 128;
    const int n0   = blockIdx.x * 128;
    const int z    = blockIdx.z;
    const __bf16* Xb = X + (size_t)z * KDIM * Ntot;

    v8f acc[4][2];
#pragma unroll
    for (int i = 0; i < 4; ++i)
#pragma unroll
        for (int j = 0; j < 2; ++j)
#pragma unroll
            for (int r = 0; r < 8; ++r) acc[i][j][r] = 0.f;

    const int khalf = lane >> 4;
    const int l16   = lane & 15;
    const int nl    = tid & 127;     // X staging column
    const int kb    = tid >> 7;      // X staging k phase (0/1)
    const int wr0   = tid >> 2;      // W staging row (chunks 0..255)
    const int wq0   = tid & 3;       // W staging 8-elem chunk in row
    const int wr1   = (tid + 256) >> 2;
    const int wq1   = (tid + 256) & 3;

    const int nsteps = KDIM / 32;

    // ---- prologue: stage tiles for k-step 0 ----
    async_ld_b128(lds_off32(&Wl[0][wr0 * 40 + wq0 * 8]),
                  Wt + (size_t)(m0 + wr0) * KDIM + wq0 * 8);
    async_ld_b128(lds_off32(&Wl[0][wr1 * 40 + wq1 * 8]),
                  Wt + (size_t)(m0 + wr1) * KDIM + wq1 * 8);
#pragma unroll
    for (int i = 0; i < 16; ++i) {
        int k = kb + i * 2;
        Xl[0][nl * 40 + k] = Xb[(size_t)k * Ntot + n0 + nl];
    }
    wait_async0();
    __syncthreads();

    for (int s = 0; s < nsteps; ++s) {
        const int buf = s & 1;
        const int nxt = buf ^ 1;
        const bool pf = (s + 1 < nsteps);

        // prefetch next tile: async W -> LDS, X -> VGPRs
        __bf16 xr[16];
        if (pf) {
            const int k0n = (s + 1) * 32;
            async_ld_b128(lds_off32(&Wl[nxt][wr0 * 40 + wq0 * 8]),
                          Wt + (size_t)(m0 + wr0) * KDIM + k0n + wq0 * 8);
            async_ld_b128(lds_off32(&Wl[nxt][wr1 * 40 + wq1 * 8]),
                          Wt + (size_t)(m0 + wr1) * KDIM + k0n + wq1 * 8);
#pragma unroll
            for (int i = 0; i < 16; ++i)
                xr[i] = Xb[(size_t)(k0n + kb + i * 2) * Ntot + n0 + nl];
        }

        // compute on current buffer
        bf16x16 af[4], bfv[2];
#pragma unroll
        for (int mb = 0; mb < 4; ++mb)
            af[mb] = load_a_frag(&Wl[buf][(wm * 64 + mb * 16 + l16) * 40], khalf);
#pragma unroll
        for (int nb = 0; nb < 2; ++nb)
            bfv[nb] = load_b_frag(&Xl[buf][(wn * 32 + nb * 16 + l16) * 40], khalf);
#pragma unroll
        for (int mb = 0; mb < 4; ++mb)
#pragma unroll
            for (int nb = 0; nb < 2; ++nb)
                acc[mb][nb] = wmma_bf16(af[mb], bfv[nb], acc[mb][nb]);

        // drain X prefetch into next buffer
        if (pf) {
#pragma unroll
            for (int i = 0; i < 16; ++i)
                Xl[nxt][nl * 40 + kb + i * 2] = xr[i];
        }
        wait_async0();
        __syncthreads();
    }

    const size_t outbase = (size_t)z * Mtot * Ntot;
#pragma unroll
    for (int mb = 0; mb < 4; ++mb)
#pragma unroll
        for (int nb = 0; nb < 2; ++nb)
#pragma unroll
            for (int r = 0; r < 8; ++r) {
                int mg = m0 + wm * 64 + mb * 16 + r + 8 * khalf;
                int ng = n0 + wn * 32 + nb * 16 + l16;
                size_t idx = outbase + (size_t)mg * Ntot + ng;
                float v = acc[mb][nb][r];
                if (MODE == 0) {
                    out16[idx] = (__bf16)v;
                } else if (MODE == 1) {
                    v += shift[mg] + resid[idx];
                    out32[idx] = v;
                    out16[idx] = (__bf16)v;
                } else if (MODE == 2) {
                    v = v * scale[mg] + shift[mg];
                    v = v / (1.f + __expf(-v));  // SiLU
                    out16[idx] = (__bf16)v;
                } else {
                    v = v * scale[mg] + shift[mg] + resid[idx];
                    out32[idx] = v;
                }
            }
}

// ---------------- offset branch: depthwise 6x6 s4 + GELU + 1x1 + tanh ----------------
__global__ __launch_bounds__(256) void offsets_kernel(
    const __bf16* __restrict__ qb, const float* __restrict__ odw,
    const float* __restrict__ odb, const float* __restrict__ opw,
    float* __restrict__ coords /* [BG][2][PP] */) {
    int gwave = (blockIdx.x * 256 + threadIdx.x) >> 5;  // 0 .. BG*PP-1
    int lane  = threadIdx.x & 31;
    int bg = gwave / PP;
    int p  = gwave % PP;
    int b  = bg / GROUPS;
    int g  = bg % GROUPS;
    int py = p / WP;
    int px = p % WP;

    float s0 = 0.f, s1 = 0.f;
#pragma unroll
    for (int cc = 0; cc < 2; ++cc) {
        int ch = lane * 2 + cc;  // 0..63
        float a = odb[ch];
        const __bf16* qrow = qb + ((size_t)b * CC + g * GD + ch) * HWSZ;
        const float*  wrow = odw + ch * (KOFF * KOFF);
#pragma unroll
        for (int ky = 0; ky < KOFF; ++ky) {
            int iy = py * DSTR - 1 + ky;
            if (iy < 0 || iy >= HH) continue;
#pragma unroll
            for (int kx = 0; kx < KOFF; ++kx) {
                int ix = px * DSTR - 1 + kx;
                if (ix < 0 || ix >= WW) continue;
                a += (float)qrow[iy * WW + ix] * wrow[ky * KOFF + kx];
            }
        }
        // tanh-approx GELU (jax.nn.gelu default approximate=True)
        float t = 0.7978845608028654f * (a + 0.044715f * a * a * a);
        float h = 0.5f * a * (1.f + tanhf(t));
        s0 += opw[ch] * h;
        s1 += opw[GD + ch] * h;
    }
#pragma unroll
    for (int off = 16; off >= 1; off >>= 1) {
        s0 += __shfl_xor(s0, off, 32);
        s1 += __shfl_xor(s1, off, 32);
    }
    if (lane == 0) {
        float ox = tanhf(s0) * (float)DSTR;
        float oy = tanhf(s1) * (float)DSTR;
        float gx = ox + (float)px;
        float gy = oy + (float)py;
        coords[((size_t)bg * 2 + 0) * PP + p] = 2.f * gx / (float)(WP - 1) - 1.f;
        coords[((size_t)bg * 2 + 1) * PP + p] = 2.f * gy / (float)(HP - 1) - 1.f;
    }
}

// ---------------- bilinear grid sample (zeros pad, align_corners=False) ----------------
__global__ __launch_bounds__(256) void sample_kernel(
    const float* __restrict__ x, const float* __restrict__ coords,
    __bf16* __restrict__ kvb /* [B][C][P] */) {
    int t = blockIdx.x * 256 + threadIdx.x;  // 0 .. B*C*P-1
    int p = t & 255;
    int c = (t >> 8) & 255;
    int b = t >> 16;
    int g = c >> 6;
    int bg = b * GROUPS + g;
    float nx = coords[((size_t)bg * 2 + 0) * PP + p];
    float ny = coords[((size_t)bg * 2 + 1) * PP + p];
    float px = (nx + 1.f) * (0.5f * WW) - 0.5f;
    float py = (ny + 1.f) * (0.5f * HH) - 0.5f;
    float x0f = floorf(px), y0f = floorf(py);
    int x0 = (int)x0f, y0 = (int)y0f;
    float wx = px - x0f, wy = py - y0f;
    const float* feat = x + ((size_t)b * CC + c) * HWSZ;
    int x1 = x0 + 1, y1 = y0 + 1;
    float v00 = (y0 >= 0 && y0 < HH && x0 >= 0 && x0 < WW) ? feat[y0 * WW + x0] : 0.f;
    float v01 = (y0 >= 0 && y0 < HH && x1 >= 0 && x1 < WW) ? feat[y0 * WW + x1] : 0.f;
    float v10 = (y1 >= 0 && y1 < HH && x0 >= 0 && x0 < WW) ? feat[y1 * WW + x0] : 0.f;
    float v11 = (y1 >= 0 && y1 < HH && x1 >= 0 && x1 < WW) ? feat[y1 * WW + x1] : 0.f;
    float val = v00 * (1.f - wx) * (1.f - wy) + v01 * wx * (1.f - wy) +
                v10 * (1.f - wx) * wy + v11 * wx * wy;
    kvb[((size_t)b * CC + c) * PP + p] = (__bf16)val;
}

// ---------------- attention core ----------------
// block = (b, head, 128-query tile); 8 waves, wave w owns query rows w*16..w*16+15.
__global__ __launch_bounds__(256) void attn_kernel(
    const __bf16* __restrict__ qb, const __bf16* __restrict__ kb,
    const __bf16* __restrict__ vb, __bf16* __restrict__ ob) {
    __shared__ __align__(16) __bf16 bufA[128 * 72];   // qT then kT chunks   (18.4 KB)
    __shared__ __align__(16) __bf16 bufV[64 * 136];   // v chunks / out tile (17.4 KB)
    __shared__ __align__(16) __bf16 scr[8 * 16 * 40]; // per-wave attn A staging (10.2 KB)

    const int tid   = threadIdx.x;
    const int lane  = tid & 31;
    const int w     = tid >> 5;
    const int khalf = lane >> 4;
    const int l16   = lane & 15;
    const int b  = blockIdx.z;
    const int hd = blockIdx.y;
    const int i0 = blockIdx.x * 128;
    const size_t qoff = ((size_t)b * CC + hd * DHEAD) * HWSZ;
    const size_t koff = ((size_t)b * CC + hd * DHEAD) * PP;

    // stage qT[i][d] (transpose: thread-copy path)
    {
        int il = tid & 127, db = tid >> 7;
#pragma unroll
        for (int i = 0; i < 32; ++i) {
            int d = db + i * 2;
            bufA[il * 72 + d] = qb[qoff + (size_t)d * HWSZ + i0 + il];
        }
    }
    __syncthreads();
    bf16x16 aq[2];
#pragma unroll
    for (int ks = 0; ks < 2; ++ks)
        aq[ks] = load_a_frag(bufA + (w * 16 + l16) * 72 + ks * 32, khalf);
    __syncthreads();

    // sim accumulation: 16 n-blocks of 16 keys
    v8f sim[16];
#pragma unroll
    for (int f = 0; f < 16; ++f)
#pragma unroll
        for (int r = 0; r < 8; ++r) sim[f][r] = 0.f;

    for (int pc = 0; pc < 2; ++pc) {
        {
            int pl = tid & 127, db = tid >> 7;
#pragma unroll
            for (int i = 0; i < 32; ++i) {
                int d = db + i * 2;
                bufA[pl * 72 + d] = kb[koff + (size_t)d * PP + pc * 128 + pl];
            }
        }
        __syncthreads();
#pragma unroll
        for (int nb = 0; nb < 8; ++nb)
#pragma unroll
            for (int ks = 0; ks < 2; ++ks) {
                bf16x16 bk = load_b_frag(bufA + (nb * 16 + l16) * 72 + ks * 32, khalf);
                sim[pc * 8 + nb] = wmma_bf16(aq[ks], bk, sim[pc * 8 + nb]);
            }
        __syncthreads();
    }

    // softmax over 256 keys per query row
    float rmax[8], rsum[8];
#pragma unroll
    for (int r = 0; r < 8; ++r) rmax[r] = -1e30f;
#pragma unroll
    for (int f = 0; f < 16; ++f)
#pragma unroll
        for (int r = 0; r < 8; ++r) {
            sim[f][r] *= 0.125f;  // DH^-0.5
            rmax[r] = fmaxf(rmax[r], sim[f][r]);
        }
#pragma unroll
    for (int r = 0; r < 8; ++r)
#pragma unroll
        for (int off = 8; off >= 1; off >>= 1)
            rmax[r] = fmaxf(rmax[r], __shfl_xor(rmax[r], off, 32));
#pragma unroll
    for (int r = 0; r < 8; ++r) rsum[r] = 0.f;
#pragma unroll
    for (int f = 0; f < 16; ++f)
#pragma unroll
        for (int r = 0; r < 8; ++r) {
            sim[f][r] = __expf(sim[f][r] - rmax[r]);
            rsum[r] += sim[f][r];
        }
#pragma unroll
    for (int r = 0; r < 8; ++r)
#pragma unroll
        for (int off = 8; off >= 1; off >>= 1)
            rsum[r] += __shfl_xor(rsum[r], off, 32);
#pragma unroll
    for (int r = 0; r < 8; ++r) {
        float inv = 1.f / rsum[r];
#pragma unroll
        for (int f = 0; f < 16; ++f) sim[f][r] *= inv;
    }

    // out = attn @ v^T  (M=16 queries, N=64 dims, K=256 keys)
    v8f oacc[4];
#pragma unroll
    for (int nd = 0; nd < 4; ++nd)
#pragma unroll
        for (int r = 0; r < 8; ++r) oacc[nd][r] = 0.f;

    __bf16* myscr = scr + w * 16 * 40;
    const int vseg = tid & 15;   // 8-elem chunk within 128-wide row
    const int vdr  = tid >> 4;   // base d row (0..15)
    for (int pc = 0; pc < 2; ++pc) {
        // v chunk: layout-preserving copy -> async global->LDS
#pragma unroll
        for (int i = 0; i < 4; ++i) {
            int d = vdr + i * 16;
            async_ld_b128(lds_off32(&bufV[d * 136 + vseg * 8]),
                          vb + koff + (size_t)d * PP + pc * 128 + vseg * 8);
        }
        wait_async0();
        __syncthreads();
#pragma unroll
        for (int kc = 0; kc < 4; ++kc) {
            // re-layout attn D-frags -> A-fragment staging (wave-private, in-order LDS)
#pragma unroll
            for (int half = 0; half < 2; ++half) {
                int f = pc * 8 + kc * 2 + half;
#pragma unroll
                for (int r = 0; r < 8; ++r)
                    myscr[(r + 8 * khalf) * 40 + half * 16 + l16] = (__bf16)sim[f][r];
            }
            bf16x16 aa = load_a_frag(myscr + l16 * 40, khalf);
#pragma unroll
            for (int nd = 0; nd < 4; ++nd) {
                bf16x16 bv = load_b_frag(bufV + (nd * 16 + l16) * 136 + kc * 32, khalf);
                oacc[nd] = wmma_bf16(aa, bv, oacc[nd]);
            }
        }
        __syncthreads();
    }

    // transpose out frags through LDS -> async coalesced store [d][i]
#pragma unroll
    for (int nd = 0; nd < 4; ++nd)
#pragma unroll
        for (int r = 0; r < 8; ++r)
            bufV[(nd * 16 + l16) * 136 + w * 16 + r + 8 * khalf] = (__bf16)oacc[nd][r];
    __syncthreads();
#pragma unroll
    for (int i = 0; i < 4; ++i) {
        int d = vdr + i * 16;
        async_st_b128(ob + qoff + (size_t)d * HWSZ + i0 + vseg * 8,
                      lds_off32(&bufV[d * 136 + vseg * 8]));
    }
    // ASYNCcnt drained by s_endpgm implicit wait-idle
}

// ---------------- host-side orchestration ----------------
extern "C" void kernel_launch(void* const* d_in, const int* in_sizes, int n_in,
                              void* d_out, int out_size, void* d_ws, size_t ws_size,
                              hipStream_t stream) {
    const float* x   = (const float*)d_in[0];
    const float* Wq  = (const float*)d_in[1];
    const float* odw = (const float*)d_in[2];
    const float* odb = (const float*)d_in[3];
    const float* opw = (const float*)d_in[4];
    const float* Wk  = (const float*)d_in[5];
    const float* Wv  = (const float*)d_in[6];
    const float* Wo  = (const float*)d_in[7];
    const float* bo  = (const float*)d_in[8];
    const float* f1w = (const float*)d_in[9];
    const float* f1g = (const float*)d_in[10];
    const float* f1b = (const float*)d_in[11];
    const float* f1m = (const float*)d_in[12];
    const float* f1v = (const float*)d_in[13];
    const float* f2w = (const float*)d_in[14];
    const float* f2g = (const float*)d_in[15];
    const float* f2b = (const float*)d_in[16];
    const float* f2m = (const float*)d_in[17];
    const float* f2v = (const float*)d_in[18];
    float* out = (float*)d_out;

    // workspace carving (256B aligned)
    char* ws = (char*)d_ws;
    size_t off = 0;
    auto carve = [&](size_t bytes) -> char* {
        char* p = ws + off;
        off += (bytes + 255) & ~(size_t)255;
        return p;
    };
    __bf16* xb    = (__bf16*)carve((size_t)NELEM * 2);
    __bf16* qbuf  = (__bf16*)carve((size_t)NELEM * 2);
    __bf16* aob   = (__bf16*)carve((size_t)NELEM * 2);   // attention output (pre-Wo)
    __bf16* a16   = (__bf16*)carve((size_t)NELEM * 2);
    float*  a32   = (float*)carve((size_t)NELEM * 4);
    __bf16* y1    = (__bf16*)carve((size_t)2 * NELEM);   // [B][2C][HW] bf16
    __bf16* Wq_b  = (__bf16*)carve((size_t)CC * CC * 2);
    __bf16* Wk_b  = (__bf16*)carve((size_t)CC * CC * 2);
    __bf16* Wv_b  = (__bf16*)carve((size_t)CC * CC * 2);
    __bf16* Wo_b  = (__bf16*)carve((size_t)CC * CC * 2);
    __bf16* f1_wb = (__bf16*)carve((size_t)2 * CC * CC * 2);
    __bf16* f2_wb = (__bf16*)carve((size_t)2 * CC * CC * 2);
    float*  coords = (float*)carve((size_t)BG * 2 * PP * 4);
    __bf16* kvb   = (__bf16*)carve((size_t)BB * CC * PP * 2);
    __bf16* kbuf  = (__bf16*)carve((size_t)BB * CC * PP * 2);
    __bf16* vbuf  = (__bf16*)carve((size_t)BB * CC * PP * 2);
    float*  s1    = (float*)carve(2 * CC * 4);
    float*  sh1   = (float*)carve(2 * CC * 4);
    float*  s2    = (float*)carve(CC * 4);
    float*  sh2   = (float*)carve(CC * 4);
    (void)ws_size; (void)in_sizes; (void)n_in; (void)out_size;

    // 1) converts + BN prep
    cvt_f32_bf16_kernel<<<4096, 256, 0, stream>>>(x, xb, NELEM);
    cvt_f32_bf16_kernel<<<256, 256, 0, stream>>>(Wq, Wq_b, CC * CC);
    cvt_f32_bf16_kernel<<<256, 256, 0, stream>>>(Wk, Wk_b, CC * CC);
    cvt_f32_bf16_kernel<<<256, 256, 0, stream>>>(Wv, Wv_b, CC * CC);
    cvt_f32_bf16_kernel<<<256, 256, 0, stream>>>(Wo, Wo_b, CC * CC);
    cvt_f32_bf16_kernel<<<512, 256, 0, stream>>>(f1w, f1_wb, 2 * CC * CC);
    cvt_f32_bf16_kernel<<<512, 256, 0, stream>>>(f2w, f2_wb, 2 * CC * CC);
    bn_prep_kernel<<<2, 256, 0, stream>>>(f1g, f1b, f1m, f1v, f2g, f2b, f2m, f2v,
                                          s1, sh1, s2, sh2);

    // 2) q = Wq @ x
    gemm_bf16_kernel<0><<<dim3(HWSZ / 128, CC / 128, BB), 256, 0, stream>>>(
        Wq_b, xb, nullptr, qbuf, nullptr, nullptr, nullptr, CC, HWSZ, CC);

    // 3) offsets -> normalized sampling coords
    offsets_kernel<<<(BG * PP) / 8, 256, 0, stream>>>(qbuf, odw, odb, opw, coords);

    // 4) bilinear grid sample -> kv (bf16)
    sample_kernel<<<(BB * CC * PP) / 256, 256, 0, stream>>>(x, coords, kvb);

    // 5) k = Wk @ kv ; v = Wv @ kv
    gemm_bf16_kernel<0><<<dim3(PP / 128, CC / 128, BB), 256, 0, stream>>>(
        Wk_b, kvb, nullptr, kbuf, nullptr, nullptr, nullptr, CC, PP, CC);
    gemm_bf16_kernel<0><<<dim3(PP / 128, CC / 128, BB), 256, 0, stream>>>(
        Wv_b, kvb, nullptr, vbuf, nullptr, nullptr, nullptr, CC, PP, CC);

    // 6) attention core
    attn_kernel<<<dim3(HWSZ / 128, HEADS, BB), 256, 0, stream>>>(qbuf, kbuf, vbuf, aob);

    // 7) a = x + Wo @ attn_out + bo   (fp32 + bf16 copies)
    gemm_bf16_kernel<1><<<dim3(HWSZ / 128, CC / 128, BB), 256, 0, stream>>>(
        Wo_b, aob, a32, a16, nullptr, bo, x, CC, HWSZ, CC);

    // 8) y1 = silu(bn1(f1_w @ a))
    gemm_bf16_kernel<2><<<dim3(HWSZ / 128, (2 * CC) / 128, BB), 256, 0, stream>>>(
        f1_wb, a16, nullptr, y1, s1, sh1, nullptr, 2 * CC, HWSZ, CC);

    // 9) out = a + bn2(f2_w @ y1)
    gemm_bf16_kernel<3><<<dim3(HWSZ / 128, CC / 128, BB), 256, 0, stream>>>(
        f2_wb, y1, out, nullptr, s2, sh2, a32, CC, HWSZ, 2 * CC);
}
